// MultiheadAttention_22926535426612
// MI455X (gfx1250) — matrix operand order repaired and checked by reading, not compiled, not run on hardware
//
#include <hip/hip_runtime.h>
#include <stdint.h>

// Problem constants (B=2, S=1024, D=1024, H=16, DK=64)
#define B_ 2
#define S_ 1024
#define D_ 1024
#define H_ 16
#define DK_ 64

typedef __attribute__((ext_vector_type(16))) __bf16 v16bf;
typedef __attribute__((ext_vector_type(8)))  float  v8f;
typedef __attribute__((ext_vector_type(4)))  unsigned int v4u;
typedef __attribute__((ext_vector_type(8)))  int v8i;
typedef __attribute__((ext_vector_type(4)))  int v4i;
typedef unsigned short u16t;
typedef unsigned int   u32t;

// fp32 -> bf16 round-to-nearest-even (bit trick; inputs are ~N(0,1))
__device__ __forceinline__ u16t f2bf(float f) {
  u32t u = __float_as_uint(f);
  u += 0x7FFFu + ((u >> 16) & 1u);
  return (u16t)(u >> 16);
}

union FragBF { v16bf v; u32t u[8]; };

// A-fragment 16x32 bf16 from row-major (row stride in elems).
// lanes 0-15 row m, K {0..7,16..23}; lanes 16-31 row m, K {8..15,24..31}.
__device__ __forceinline__ v16bf load_frag_a(const u16t* base, int stride, int lane) {
  int m = lane & 15, half = lane >> 4;
  const u16t* rp = base + (size_t)m * stride + 8 * half;
  FragBF f;
#pragma unroll
  for (int p = 0; p < 8; ++p) {
    int k0 = 2 * (p & 3) + ((p & 4) << 2);  // 0,2,4,6,16,18,20,22
    f.u[p] = *reinterpret_cast<const u32t*>(rp + k0);
  }
  return f.v;
}

// B-fragment 32x16 bf16, staged as B^T (N x K row-major).
// lanes 0-15: col n, K {0..7,16..23}; lanes 16-31: col n, K {8..15,24..31}.
__device__ __forceinline__ v16bf load_frag_bT(const u16t* baseT, int stride, int lane) {
  int n = lane & 15, half = lane >> 4;
  const u16t* rp = baseT + (size_t)n * stride + 8 * half;
  FragBF f;
#pragma unroll
  for (int p = 0; p < 8; ++p)
    f.u[p] = *reinterpret_cast<const u32t*>(rp + 2 * (p & 3) + ((p & 4) << 2));
  return f.v;
}

#define WMMA_BF16(a, b, c) \
  __builtin_amdgcn_wmma_f32_16x16x32_bf16(false, (a), false, (b), (short)0, (c), false, false)

__device__ __forceinline__ float red_max16(float x) {
#pragma unroll
  for (int m = 1; m <= 8; m <<= 1) x = fmaxf(x, __shfl_xor(x, m, 32));
  return x;
}
__device__ __forceinline__ float red_sum16(float x) {
#pragma unroll
  for (int m = 1; m <= 8; m <<= 1) x += __shfl_xor(x, m, 32);
  return x;
}

// ---- Tensor Data Mover: 2D tile DMA global -> LDS (bf16 elements) ----
// D# per CDNA5 ISA. group0: count=1, lds_addr[63:32],
// global_addr[120:64], type[127:126]=2. group1: data_size=1 (2B),
// tensor_dim0[79:48], tensor_dim1[111:80], tile_dim0[127:112],
// tile_dim1[143:128], tensor_dim0_stride[207:160].
__device__ __forceinline__ u32t lds_byte_off(const void* p) {
  // flat shared addr = {aperture[63:32], lds_offset[31:0]} (ISA §10.2)
  return (u32t)(uintptr_t)p;
}

__device__ __forceinline__ void tdm_load_2d(const void* lds_dst, const void* gsrc,
                                            u32t tile_d0, u32t tile_d1,
                                            u32t stride0_elems) {
  uint64_t ga = (uint64_t)(uintptr_t)gsrc;
  union { v4u v; u32t u[4]; } g0;
  g0.u[0] = 1u;                                            // count=1 (user D#)
  g0.u[1] = lds_byte_off(lds_dst);                         // lds_addr (bytes)
  g0.u[2] = (u32t)(ga & 0xFFFFFFFFu);                      // global_addr[95:64]
  g0.u[3] = (u32t)((ga >> 32) & 0x01FFFFFFu) | (2u << 30); // addr[120:96] | type=2
  union { v8i v; u32t u[8]; } g1;
  u32t td0 = tile_d0, td1 = tile_d1;                       // tensor dims == tile (in-bounds)
  g1.u[0] = (1u << 16);                                    // wg_mask=0, data_size=1 (2B)
  g1.u[1] = (td0 & 0xFFFFu) << 16;                         // tensor_dim0[15:0]
  g1.u[2] = ((td0 >> 16) & 0xFFFFu) | ((td1 & 0xFFFFu) << 16);
  g1.u[3] = ((td1 >> 16) & 0xFFFFu) | ((tile_d0 & 0xFFFFu) << 16);
  g1.u[4] = tile_d1 & 0xFFFFu;                             // tile_dim1, tile_dim2=0
  g1.u[5] = stride0_elems;                                 // tensor_dim0_stride[31:0]
  g1.u[6] = 0;                                             // stride0 hi, stride1 lo
  g1.u[7] = 0;
  v4i z4 = (v4i){};
  v8i z8 = (v8i){};
  __builtin_amdgcn_tensor_load_to_lds(g0.v, g1.v, z4, z4, z8, 0);
}

// ---------------- Stage 0: conversions ----------------
__global__ void cvt_f32_bf16(const float* __restrict__ in, u16t* __restrict__ out, int n) {
  for (int i = blockIdx.x * blockDim.x + threadIdx.x; i < n; i += gridDim.x * blockDim.x)
    out[i] = f2bf(in[i]);
}

// out[bi][c][r] = in[bi][r][c]  (convert + transpose; output-coalesced)
__global__ void tcvt_f32_bf16(const float* __restrict__ in, u16t* __restrict__ out,
                              int rows, int cols, int n) {
  for (int o = blockIdx.x * blockDim.x + threadIdx.x; o < n; o += gridDim.x * blockDim.x) {
    int r = o % rows;
    int c = (o / rows) % cols;
    int bi = o / (rows * cols);
    out[o] = f2bf(in[(size_t)bi * rows * cols + (size_t)r * cols + c]);
  }
}

// ---------------- Stage 1: per-head projections ----------------
// grid (S/128, H, B*3), block 128 (4 waves). Wave owns 32 rows x 64 cols
// (2 A-frags, 8 accum tiles -> 8 WMMA per 32-K step). Ping-pong K-loop
// (unroll x2) so prefetched fragments need no register copies.
__global__ __launch_bounds__(128) void proj_kernel(
    const u16t* __restrict__ xq, const u16t* __restrict__ xk, const u16t* __restrict__ xv,
    const u16t* __restrict__ WqT, const u16t* __restrict__ WkT, const u16t* __restrict__ WvT,
    u16t* __restrict__ qh, u16t* __restrict__ kh, u16t* __restrict__ vT) {
  int lane = threadIdx.x & 31, wave = threadIdx.x >> 5;
  int z = blockIdx.z, b = z / 3, proj = z % 3;
  int h = blockIdx.y;
  int s0 = blockIdx.x * 128 + wave * 32;

  const u16t* x  = (proj == 0) ? xq  : (proj == 1) ? xk  : xv;
  const u16t* WT = (proj == 0) ? WqT : (proj == 1) ? WkT : WvT;
  const u16t* xb  = x + ((size_t)b * S_ + s0) * D_;
  const u16t* WTh = WT + (size_t)h * DK_ * D_;

  v8f acc[2][4];
#pragma unroll
  for (int i = 0; i < 2; ++i)
#pragma unroll
    for (int j = 0; j < 4; ++j) acc[i][j] = (v8f){};

  v16bf A0[2], B0[4], A1[2], B1[4];
#pragma unroll
  for (int i = 0; i < 2; ++i) A0[i] = load_frag_a(xb + (size_t)(16 * i) * D_, D_, lane);
#pragma unroll
  for (int j = 0; j < 4; ++j) B0[j] = load_frag_bT(WTh + (size_t)(16 * j) * D_, D_, lane);

  for (int kk = 0; kk < D_; kk += 64) {
    int k1 = kk + 32;
#pragma unroll
    for (int i = 0; i < 2; ++i) A1[i] = load_frag_a(xb + (size_t)(16 * i) * D_ + k1, D_, lane);
#pragma unroll
    for (int j = 0; j < 4; ++j) B1[j] = load_frag_bT(WTh + (size_t)(16 * j) * D_ + k1, D_, lane);
#pragma unroll
    for (int i = 0; i < 2; ++i)
#pragma unroll
      for (int j = 0; j < 4; ++j) acc[i][j] = WMMA_BF16(A0[i], B0[j], acc[i][j]);

    int k2 = (kk + 64 < D_) ? kk + 64 : 0;  // last iter harmlessly re-reads tile 0
#pragma unroll
    for (int i = 0; i < 2; ++i) A0[i] = load_frag_a(xb + (size_t)(16 * i) * D_ + k2, D_, lane);
#pragma unroll
    for (int j = 0; j < 4; ++j) B0[j] = load_frag_bT(WTh + (size_t)(16 * j) * D_ + k2, D_, lane);
#pragma unroll
    for (int i = 0; i < 2; ++i)
#pragma unroll
      for (int j = 0; j < 4; ++j) acc[i][j] = WMMA_BF16(A1[i], B1[j], acc[i][j]);
  }

  int n = lane & 15, half = lane >> 4;
  if (proj < 2) {
    u16t* outb = ((proj == 0) ? qh : kh) + (((size_t)b * H_ + h) * S_ + s0) * DK_;
#pragma unroll
    for (int i = 0; i < 2; ++i)
#pragma unroll
      for (int j = 0; j < 4; ++j)
#pragma unroll
        for (int r = 0; r < 8; ++r)
          outb[(size_t)(16 * i + r + 8 * half) * DK_ + 16 * j + n] = f2bf(acc[i][j][r]);
  } else {
    // store V transposed: vT[b,h,dk,s]
    u16t* outb = vT + ((size_t)b * H_ + h) * DK_ * S_;
#pragma unroll
    for (int i = 0; i < 2; ++i)
#pragma unroll
      for (int j = 0; j < 4; ++j)
#pragma unroll
        for (int r = 0; r < 8; ++r)
          outb[(size_t)(16 * j + n) * S_ + s0 + 16 * i + r + 8 * half] = f2bf(acc[i][j][r]);
  }
}

// ---------------- Stage 2: flash attention with TDM-staged K/V ----------------
// grid (S/64, H, B), block 128. Wave owns 16 query rows, DK=64 output cols.
// Wave 0 drives the Tensor Data Mover: double-buffered K (32x64) and V^T (64x32)
// chunks in LDS, DMA of chunk i+1 overlapped with compute of chunk i.
__global__ __launch_bounds__(128) void attn_kernel(
    const u16t* __restrict__ qh, const u16t* __restrict__ kh, const u16t* __restrict__ vT,
    const unsigned char* __restrict__ mask, u16t* __restrict__ concat) {
  __shared__ u16t kbuf[2][32 * 64];   // [t_local][dk]
  __shared__ u16t vbuf[2][64 * 32];   // [dk][t_local]
  __shared__ u16t pbuf[4 * 16 * 32];  // per-wave P re-layout buffer

  int lane = threadIdx.x & 31, wave = threadIdx.x >> 5;
  int b = blockIdx.z, h = blockIdx.y;
  int s0 = blockIdx.x * 64 + wave * 16;
  int n = lane & 15, half = lane >> 4;

  const u16t* qhb = qh + (((size_t)b * H_ + h) * S_ + s0) * DK_;
  const u16t* khb = kh + ((size_t)b * H_ + h) * S_ * DK_;
  const u16t* vTb = vT + ((size_t)b * H_ + h) * DK_ * S_;
  const unsigned char* mb = mask + (size_t)b * S_ * S_;
  u16t* pw = pbuf + wave * (16 * 32);

  v16bf aq0 = load_frag_a(qhb, DK_, lane);
  v16bf aq1 = load_frag_a(qhb + 32, DK_, lane);

  v8f O[4];
#pragma unroll
  for (int j = 0; j < 4; ++j) O[j] = (v8f){};
  float Mrow[8], Lrow[8];
#pragma unroll
  for (int r = 0; r < 8; ++r) { Mrow[r] = -1e30f; Lrow[r] = 0.0f; }

  const float scale = 0.125f;  // 1/sqrt(64)
  const int NCH = S_ / 32;

  // prologue: DMA chunk 0
  if (wave == 0) {
    tdm_load_2d(kbuf[0], khb, /*tile_d0=*/64, /*tile_d1=*/32, /*stride0=*/DK_);
    tdm_load_2d(vbuf[0], vTb, /*tile_d0=*/32, /*tile_d1=*/64, /*stride0=*/S_);
    __builtin_amdgcn_s_wait_tensorcnt(0);
  }
  __syncthreads();

  for (int ic = 0; ic < NCH; ++ic) {
    int tb = ic * 32;
    int cur = ic & 1, nxt = cur ^ 1;
    if (wave == 0 && ic + 1 < NCH) {  // kick off next chunk's DMA (overlap with compute)
      int nb = (ic + 1) * 32;
      tdm_load_2d(kbuf[nxt], khb + (size_t)nb * DK_, 64, 32, DK_);
      tdm_load_2d(vbuf[nxt], vTb + nb, 32, 64, S_);
    }
    const u16t* kc = kbuf[cur];
    const u16t* vc = vbuf[cur];

    // scores: 16 x 32 block, two 16x16 tiles, K = DK = 64 (2 WMMA each)
    float sA[2][8];
#pragma unroll
    for (int j = 0; j < 2; ++j) {
      const u16t* kbT = kc + (size_t)(16 * j) * DK_;
      v16bf b0 = load_frag_bT(kbT, DK_, lane);
      v16bf b1 = load_frag_bT(kbT + 32, DK_, lane);
      v8f c = (v8f){};
      c = WMMA_BF16(aq0, b0, c);
      c = WMMA_BF16(aq1, b1, c);
      int t = tb + 16 * j + n;
#pragma unroll
      for (int r = 0; r < 8; ++r) {
        float s = c[r] * scale;
        int srow = s0 + r + 8 * half;
        if (!mb[(size_t)srow * S_ + t]) s = -1e9f;
        sA[j][r] = s;
      }
    }
    // online softmax row stats (rows = vgpr index, cols across 16 lanes)
    float Mn[8], sc[8];
#pragma unroll
    for (int r = 0; r < 8; ++r) {
      float mt = red_max16(fmaxf(sA[0][r], sA[1][r]));
      Mn[r] = fmaxf(Mrow[r], mt);
      sc[r] = __expf(Mrow[r] - Mn[r]);
      Mrow[r] = Mn[r];
      sA[0][r] = __expf(sA[0][r] - Mn[r]);
      sA[1][r] = __expf(sA[1][r] - Mn[r]);
      float rs = red_sum16(sA[0][r] + sA[1][r]);
      Lrow[r] = Lrow[r] * sc[r] + rs;
    }
#pragma unroll
    for (int j = 0; j < 4; ++j)
#pragma unroll
      for (int r = 0; r < 8; ++r) O[j][r] *= sc[r];

    // re-layout P (C-layout -> A-layout) through per-wave LDS
#pragma unroll
    for (int r = 0; r < 8; ++r) {
      int m = r + 8 * half;
      pw[m * 32 + n]      = f2bf(sA[0][r]);
      pw[m * 32 + 16 + n] = f2bf(sA[1][r]);
    }
    asm volatile("s_wait_dscnt 0" ::: "memory");
    v16bf ap = load_frag_a(pw, 32, lane);

    // O += P (16x32) x V_chunk (32x64)
#pragma unroll
    for (int j = 0; j < 4; ++j) {
      v16bf bv = load_frag_bT(vc + (size_t)(16 * j) * 32, 32, lane);
      O[j] = WMMA_BF16(ap, bv, O[j]);
    }

    if (wave == 0) __builtin_amdgcn_s_wait_tensorcnt(0);  // next chunk landed
    __syncthreads();                                      // publish LDS, retire reads
  }

  // normalize; write head outputs directly in concat layout [b, s, h*DK+dk]
#pragma unroll
  for (int j = 0; j < 4; ++j)
#pragma unroll
    for (int r = 0; r < 8; ++r) {
      int srow = s0 + r + 8 * half;
      float o = O[j][r] / Lrow[r];
      concat[((size_t)b * S_ + srow) * D_ + h * DK_ + 16 * j + n] = f2bf(o);
    }
}

// ---------------- Stage 3: unify GEMM + bias ----------------
// out[row, d] = concat[row, :] @ Wu + bu ; rows = B*S = 2048.
// grid (16, 16), block 128; wave owns 32 rows x 64 cols; ping-pong K-loop.
__global__ __launch_bounds__(128) void unify_kernel(
    const u16t* __restrict__ concat, const u16t* __restrict__ WuT,
    const float* __restrict__ bu, float* __restrict__ out) {
  int lane = threadIdx.x & 31, wave = threadIdx.x >> 5;
  int m0 = blockIdx.x * 128 + wave * 32;
  int n0 = blockIdx.y * 64;
  int n = lane & 15, half = lane >> 4;

  const u16t* ab = concat + (size_t)m0 * D_;
  const u16t* wb = WuT + (size_t)n0 * D_;
  v8f acc[2][4];
#pragma unroll
  for (int i = 0; i < 2; ++i)
#pragma unroll
    for (int j = 0; j < 4; ++j) acc[i][j] = (v8f){};

  v16bf A0[2], B0[4], A1[2], B1[4];
#pragma unroll
  for (int i = 0; i < 2; ++i) A0[i] = load_frag_a(ab + (size_t)(16 * i) * D_, D_, lane);
#pragma unroll
  for (int j = 0; j < 4; ++j) B0[j] = load_frag_bT(wb + (size_t)(16 * j) * D_, D_, lane);

  for (int kk = 0; kk < D_; kk += 64) {
    int k1 = kk + 32;
#pragma unroll
    for (int i = 0; i < 2; ++i) A1[i] = load_frag_a(ab + (size_t)(16 * i) * D_ + k1, D_, lane);
#pragma unroll
    for (int j = 0; j < 4; ++j) B1[j] = load_frag_bT(wb + (size_t)(16 * j) * D_ + k1, D_, lane);
#pragma unroll
    for (int i = 0; i < 2; ++i)
#pragma unroll
      for (int j = 0; j < 4; ++j) acc[i][j] = WMMA_BF16(A0[i], B0[j], acc[i][j]);

    int k2 = (kk + 64 < D_) ? kk + 64 : 0;
#pragma unroll
    for (int i = 0; i < 2; ++i) A0[i] = load_frag_a(ab + (size_t)(16 * i) * D_ + k2, D_, lane);
#pragma unroll
    for (int j = 0; j < 4; ++j) B0[j] = load_frag_bT(wb + (size_t)(16 * j) * D_ + k2, D_, lane);
#pragma unroll
    for (int i = 0; i < 2; ++i)
#pragma unroll
      for (int j = 0; j < 4; ++j) acc[i][j] = WMMA_BF16(A1[i], B1[j], acc[i][j]);
  }
#pragma unroll
  for (int i = 0; i < 2; ++i)
#pragma unroll
    for (int j = 0; j < 4; ++j) {
      float bias = bu[n0 + 16 * j + n];
#pragma unroll
      for (int r = 0; r < 8; ++r)
        out[(size_t)(m0 + 16 * i + r + 8 * half) * D_ + n0 + 16 * j + n] = acc[i][j][r] + bias;
    }
}

// ---------------- host side ----------------
extern "C" void kernel_launch(void* const* d_in, const int* in_sizes, int n_in,
                              void* d_out, int out_size, void* d_ws, size_t ws_size,
                              hipStream_t stream) {
  const float* q  = (const float*)d_in[0];
  const float* k  = (const float*)d_in[1];
  const float* v  = (const float*)d_in[2];
  const unsigned char* mask = (const unsigned char*)d_in[3];
  const float* Wq = (const float*)d_in[4];
  const float* Wk = (const float*)d_in[5];
  const float* Wv = (const float*)d_in[6];
  const float* Wu = (const float*)d_in[7];
  const float* bu = (const float*)d_in[8];
  float* out = (float*)d_out;

  const size_t EL_X = (size_t)B_ * S_ * D_;      // 2,097,152
  const size_t EL_W = (size_t)H_ * D_ * DK_;     // 1,048,576
  const size_t EL_WU = (size_t)(H_ * DK_) * D_;  // 1,048,576
  const size_t EL_H = (size_t)B_ * H_ * S_ * DK_;// 2,097,152

  u16t* w = (u16t*)d_ws;
  u16t* qbf = w;                 u16t* kbf = qbf + EL_X;  u16t* vbf = kbf + EL_X;
  u16t* WqT = vbf + EL_X;        u16t* WkT = WqT + EL_W;  u16t* WvT = WkT + EL_W;
  u16t* WuT = WvT + EL_W;
  u16t* qh  = WuT + EL_WU;       u16t* kh  = qh + EL_H;   u16t* vT  = kh + EL_H;
  u16t* cc  = vT + EL_H;         // concat: EL_X elems

  cvt_f32_bf16<<<4096, 256, 0, stream>>>(q, qbf, (int)EL_X);
  cvt_f32_bf16<<<4096, 256, 0, stream>>>(k, kbf, (int)EL_X);
  cvt_f32_bf16<<<4096, 256, 0, stream>>>(v, vbf, (int)EL_X);
  tcvt_f32_bf16<<<4096, 256, 0, stream>>>(Wq, WqT, D_, DK_, (int)EL_W);
  tcvt_f32_bf16<<<4096, 256, 0, stream>>>(Wk, WkT, D_, DK_, (int)EL_W);
  tcvt_f32_bf16<<<4096, 256, 0, stream>>>(Wv, WvT, D_, DK_, (int)EL_W);
  tcvt_f32_bf16<<<4096, 256, 0, stream>>>(Wu, WuT, H_ * DK_, D_, (int)EL_WU);

  proj_kernel<<<dim3(S_ / 128, H_, B_ * 3), 128, 0, stream>>>(
      qbf, kbf, vbf, WqT, WkT, WvT, qh, kh, vT);

  attn_kernel<<<dim3(S_ / 64, H_, B_), 128, 0, stream>>>(qh, kh, vT, mask, cc);

  unify_kernel<<<dim3((B_ * S_) / 128, D_ / 64), 128, 0, stream>>>(cc, WuT, bu, out);
}
